// Head_selfEnc_38534446580279
// MI455X (gfx1250) — compile-verified, matrix operand order checked by reading, not run on hardware
//
#include <hip/hip_runtime.h>
#include <stdint.h>

typedef _Float16 half_t;
typedef __attribute__((ext_vector_type(16))) _Float16 v16h;
typedef __attribute__((ext_vector_type(8)))  _Float16 v8h;
typedef __attribute__((ext_vector_type(8)))  float    v8f;

#define B_  4
#define T_  4096
#define C_  512
#define H_  64
#define BT_ (B_ * T_)
#define LOG2E 1.44269504088896340736f

static __device__ inline v16h make16(v8h lo, v8h hi) {
  return __builtin_shufflevector(lo, hi, 0,1,2,3,4,5,6,7,8,9,10,11,12,13,14,15);
}
static __device__ inline v8h ld8(const half_t* p) { return *(const v8h*)p; }

static __device__ inline v8f wmma_f16(v16h a, v16h b, v8f c) {
  // D = A(16x32 f16) * B(32x16 f16) + C(16x16 f32)
  return __builtin_amdgcn_wmma_f32_16x16x32_f16(false, a, false, b, (short)0, c, false, false);
}

// CDNA5 async global->LDS copy (16 B per lane), tracked by ASYNCcnt.
static __device__ inline void async_cp16(unsigned lds_off, const half_t* g) {
  asm volatile("global_load_async_to_lds_b128 %0, %1, off"
               :: "v"(lds_off), "v"(g) : "memory");
}

// ---------------------------------------------------------------------------
// Kernel 0: W[c][h] (f32) -> WT[h][c] (f16), for Wq/Wk/Wv (blockIdx.y selects)
// ---------------------------------------------------------------------------
__global__ __launch_bounds__(256) void wt_convert_kernel(
    const float* __restrict__ Wq, const float* __restrict__ Wk,
    const float* __restrict__ Wv, half_t* __restrict__ WT) {
  int m = blockIdx.y;
  const float* W = (m == 0) ? Wq : ((m == 1) ? Wk : Wv);
  half_t* dst = WT + (size_t)m * H_ * C_;
  int idx = blockIdx.x * 256 + threadIdx.x;   // [0, H_*C_)
  int h = idx >> 9;
  int c = idx & (C_ - 1);
  dst[idx] = (half_t)W[c * H_ + h];
}

// ---------------------------------------------------------------------------
// Kernel 1: projection, one wave = 16 rows x 64 cols, K=512 via WMMA f16.
// vTranspose!=0 stores output as Y[b][h][t] (for V), else Y[row][h].
// ---------------------------------------------------------------------------
__global__ __launch_bounds__(256) void proj_kernel(
    const float* __restrict__ X, const half_t* __restrict__ WT,
    half_t* __restrict__ Y, float outscale, int vTranspose) {
  int lane = threadIdx.x & 31;
  int wv   = threadIdx.x >> 5;
  int lrow = lane & 15;
  int kb8  = (lane >> 4) * 8;   // A/B operand k-run base for this lane half
  int prow = (lane >> 4) * 8;   // C/D row base for this lane half

  int rowbase = blockIdx.x * 128 + wv * 16;
  const float* xr = X + (size_t)(rowbase + lrow) * C_;

  v8f o[4] = {v8f{}, v8f{}, v8f{}, v8f{}};

  for (int kc = 0; kc < C_; kc += 32) {
    v8f x0 = *(const v8f*)(xr + kc + kb8);
    v8f x1 = *(const v8f*)(xr + kc + kb8 + 16);
    v16h a = make16(__builtin_convertvector(x0, v8h),
                    __builtin_convertvector(x1, v8h));
#pragma unroll
    for (int nt = 0; nt < 4; ++nt) {
      const half_t* wr = WT + (size_t)(nt * 16 + lrow) * C_ + kc + kb8;
      v16h bm = make16(ld8(wr), ld8(wr + 16));
      o[nt] = wmma_f16(a, bm, o[nt]);
    }
  }

  // uniform branch hoisted OUT of the store loops (keeps stores branch-free)
  if (!vTranspose) {
    half_t* yr = Y + (size_t)(rowbase + prow) * H_ + lrow;
#pragma unroll
    for (int nt = 0; nt < 4; ++nt) {
#pragma unroll
      for (int r = 0; r < 8; ++r)
        yr[(size_t)r * H_ + nt * 16] = (half_t)(o[nt][r] * outscale);
    }
  } else {
    int bb = rowbase >> 12;             // batch (T_ = 4096)
    int t0 = (rowbase & (T_ - 1)) + prow;
#pragma unroll
    for (int nt = 0; nt < 4; ++nt) {
      half_t* yc = Y + ((size_t)bb * H_ + nt * 16 + lrow) * T_ + t0;
#pragma unroll
      for (int r = 0; r < 8; ++r)
        yc[r] = (half_t)(o[nt][r] * outscale);
    }
  }
}

// ---------------------------------------------------------------------------
// Kernel 2: flash attention, exp2-domain online softmax, double-buffered
// async K/V^T staging. One wave = 16 query rows; block = 128 rows.
// ---------------------------------------------------------------------------
__global__ __launch_bounds__(256) void attn_kernel(
    const half_t* __restrict__ qh, const half_t* __restrict__ kh,
    const half_t* __restrict__ vhT, float* __restrict__ out) {
  __shared__ alignas(16) half_t Ks [2][64][72];   // [buf][key][head], 144 B rows
  __shared__ alignas(16) half_t VsT[2][64][72];   // [buf][head][key]
  __shared__ alignas(16) half_t Pl [8][16][72];   // per-wave P tile (16x64)

  int lane = threadIdx.x & 31;
  int wv   = threadIdx.x >> 5;
  int lrow = lane & 15;
  int kb8  = (lane >> 4) * 8;
  int prow = (lane >> 4) * 8;

  int b  = blockIdx.y;
  int q0 = blockIdx.x * 128 + wv * 16;
  const half_t* qB = qh  + (size_t)b * T_ * H_;
  const half_t* kB = kh  + (size_t)b * T_ * H_;
  const half_t* vB = vhT + (size_t)b * H_ * T_;

  // Q A-operands (head dim = two K=32 chunks), kept in registers
  const half_t* qr = qB + (size_t)(q0 + lrow) * H_;
  v16h qa0 = make16(ld8(qr + kb8),      ld8(qr + kb8 + 16));
  v16h qa1 = make16(ld8(qr + 32 + kb8), ld8(qr + 48 + kb8));

  v8f o[4] = {v8f{}, v8f{}, v8f{}, v8f{}};
  float m[8], l[8];
#pragma unroll
  for (int r = 0; r < 8; ++r) { m[r] = -1e30f; l[r] = 0.0f; }

  // async-stage one 64-key chunk (K rows + V^T rows are both pure row copies)
  auto stage = [&](int chunk, int bufsel) {
    int kc = chunk * 64;
#pragma unroll
    for (int j = 0; j < 2; ++j) {
      int c   = threadIdx.x + j * 256;   // [0,512): 64 rows x 8 16B-segments
      int row = c >> 3;
      int sg  = (c & 7) * 8;
      async_cp16((unsigned)(uintptr_t)&Ks[bufsel][row][sg],
                 kB + (size_t)(kc + row) * H_ + sg);
      async_cp16((unsigned)(uintptr_t)&VsT[bufsel][row][sg],
                 vB + (size_t)row * T_ + kc + sg);
    }
  };

  stage(0, 0);
  asm volatile("s_wait_asynccnt 0x0" ::: "memory");
  __syncthreads();

  const int NCH = T_ / 64;
  for (int i = 0; i < NCH; ++i) {
    int buf = i & 1;
    if (i + 1 < NCH) stage(i + 1, buf ^ 1);   // prefetch next chunk

    // ---- scores: S(16x64) = Q @ K^T, four 16-key column groups ----
    v8f s[4];
#pragma unroll
    for (int g = 0; g < 4; ++g) {
      const half_t* kp = &Ks[buf][g * 16 + lrow][kb8];
      v16h b0 = make16(ld8(kp),      ld8(kp + 16));
      v16h b1 = make16(ld8(kp + 32), ld8(kp + 48));
      v8f acc = {};
      acc = wmma_f16(qa0, b0, acc);
      acc = wmma_f16(qa1, b1, acc);
      s[g] = acc;
    }

    // ---- online softmax in exp2 domain (log2e folded into Q scale) ----
#pragma unroll
    for (int r = 0; r < 8; ++r) {
      float a = fmaxf(fmaxf(s[0][r], s[1][r]), fmaxf(s[2][r], s[3][r]));
      a = fmaxf(a, __shfl_xor(a, 1));
      a = fmaxf(a, __shfl_xor(a, 2));
      a = fmaxf(a, __shfl_xor(a, 4));
      a = fmaxf(a, __shfl_xor(a, 8));
      float mn = fmaxf(m[r], a);
      float cf = __builtin_amdgcn_exp2f(m[r] - mn);
      m[r] = mn;
      float p0 = __builtin_amdgcn_exp2f(s[0][r] - mn);
      float p1 = __builtin_amdgcn_exp2f(s[1][r] - mn);
      float p2 = __builtin_amdgcn_exp2f(s[2][r] - mn);
      float p3 = __builtin_amdgcn_exp2f(s[3][r] - mn);
      s[0][r] = p0; s[1][r] = p1; s[2][r] = p2; s[3][r] = p3;
      float rs = (p0 + p1) + (p2 + p3);
      rs += __shfl_xor(rs, 1);
      rs += __shfl_xor(rs, 2);
      rs += __shfl_xor(rs, 4);
      rs += __shfl_xor(rs, 8);
      l[r] = l[r] * cf + rs;
      o[0][r] *= cf; o[1][r] *= cf; o[2][r] *= cf; o[3][r] *= cf;
    }

    // ---- stage P (C/D layout) -> LDS -> reread as A operand (f16) ----
#pragma unroll
    for (int r = 0; r < 8; ++r) {
#pragma unroll
      for (int g = 0; g < 4; ++g)
        Pl[wv][prow + r][g * 16 + lrow] = (half_t)s[g][r];
    }
    asm volatile("s_wait_dscnt 0x0" ::: "memory");  // same-wave LDS RAW
    const half_t* pp = &Pl[wv][lrow][0];
    v16h pa0 = make16(ld8(pp + kb8),      ld8(pp + kb8 + 16));
    v16h pa1 = make16(ld8(pp + 32 + kb8), ld8(pp + 48 + kb8));

    // ---- O += P @ V over 64 keys (two K=32 chunks x four head tiles) ----
#pragma unroll
    for (int nt = 0; nt < 4; ++nt) {
      const half_t* vp = &VsT[buf][nt * 16 + lrow][0];
      v16h vb0 = make16(ld8(vp + kb8),      ld8(vp + kb8 + 16));
      v16h vb1 = make16(ld8(vp + 32 + kb8), ld8(vp + 48 + kb8));
      o[nt] = wmma_f16(pa0, vb0, o[nt]);
      o[nt] = wmma_f16(pa1, vb1, o[nt]);
    }

    asm volatile("s_wait_asynccnt 0x0" ::: "memory"); // next tile resident
    __syncthreads();                                  // everyone done with buf
  }

  float rl[8];
#pragma unroll
  for (int r = 0; r < 8; ++r) rl[r] = __builtin_amdgcn_rcpf(l[r]);

  float* op = out + ((size_t)b * T_ + q0 + prow) * H_ + lrow;
#pragma unroll
  for (int r = 0; r < 8; ++r) {
    float* orow = op + (size_t)r * H_;
#pragma unroll
    for (int nt = 0; nt < 4; ++nt)
      orow[nt * 16] = o[nt][r] * rl[r];
  }
}

// ---------------------------------------------------------------------------
extern "C" void kernel_launch(void* const* d_in, const int* in_sizes, int n_in,
                              void* d_out, int out_size, void* d_ws, size_t ws_size,
                              hipStream_t stream) {
  const float* q  = (const float*)d_in[0];
  const float* k  = (const float*)d_in[1];
  const float* v  = (const float*)d_in[2];
  const float* Wq = (const float*)d_in[3];
  const float* Wk = (const float*)d_in[4];
  const float* Wv = (const float*)d_in[5];
  float* out = (float*)d_out;

  half_t* wsh = (half_t*)d_ws;
  half_t* qh  = wsh;                                 // [BT_][H_]   f16
  half_t* kh  = wsh + (size_t)BT_ * H_;              // [BT_][H_]   f16
  half_t* vhT = wsh + 2 * (size_t)BT_ * H_;          // [B_][H_][T_] f16
  half_t* WT  = wsh + 3 * (size_t)BT_ * H_;          // 3 x [H_][C_] f16

  wt_convert_kernel<<<dim3((C_ * H_) / 256, 3), 256, 0, stream>>>(Wq, Wk, Wv, WT);
  // fold attention scale AND log2(e) (exp2-domain softmax) into qh
  proj_kernel<<<dim3(BT_ / 128), 256, 0, stream>>>(q, WT, qh, 0.125f * LOG2E, 0);
  proj_kernel<<<dim3(BT_ / 128), 256, 0, stream>>>(k, WT + (size_t)H_ * C_, kh, 1.0f, 0);
  proj_kernel<<<dim3(BT_ / 128), 256, 0, stream>>>(v, WT + 2 * (size_t)H_ * C_, vhT, 1.0f, 1);
  attn_kernel<<<dim3(T_ / 128, B_), 256, 0, stream>>>(qh, kh, vhT, out);
}
